// MyModel_87454124082074
// MI455X (gfx1250) — matrix operand order repaired and checked by reading, not compiled, not run on hardware
//
#include <hip/hip_runtime.h>
#include <hip/hip_bf16.h>

// ---------------------------------------------------------------------------
// Types
// ---------------------------------------------------------------------------
typedef __attribute__((ext_vector_type(16))) __bf16 v16bf;
typedef __attribute__((ext_vector_type(8)))  __bf16 v8bf;
typedef __attribute__((ext_vector_type(8)))  float  v8f;

#define VOCAB 32000
#define HID   512
#define T_IN  128
#define T_OUT 128
#define BATCH 32

__device__ __forceinline__ v8f wmma_bf16(v16bf a, v16bf b, v8f c) {
    // D = A(16x32 bf16) * B(32x16 bf16) + C(16x16 f32)
    return __builtin_amdgcn_wmma_f32_16x16x32_bf16(
        /*neg_a=*/false, a, /*neg_b=*/false, b,
        /*c_mod=*/(short)0, c, /*reuse_a=*/false, /*reuse_b=*/false);
}

__device__ __forceinline__ float sigmoidf_fast(float x) {
    return 1.0f / (1.0f + __expf(-x));
}

// ---------------------------------------------------------------------------
// Weight prep: fp32 [K,N] -> bf16 transposed [N,K], 32x32 LDS tiles,
// coalesced loads AND stores. K,N multiples of 32.
// Block: (32,8) = 256 threads, each thread handles 4 tile rows.
// ---------------------------------------------------------------------------
__global__ __launch_bounds__(256) void transpose_to_bf16(
    const float* __restrict__ in, __bf16* __restrict__ out, int K, int N) {
    __shared__ float tile[32][33];
    const int tx = threadIdx.x;          // 0..31
    const int ty = threadIdx.y;          // 0..7
    const int nb = blockIdx.x * 32;      // N-tile base
    const int kb = blockIdx.y * 32;      // K-tile base

    #pragma unroll
    for (int j = 0; j < 4; ++j) {
        int k = kb + ty + j * 8;
        tile[ty + j * 8][tx] = in[(long)k * N + nb + tx];
    }
    __syncthreads();
    #pragma unroll
    for (int j = 0; j < 4; ++j) {
        int n = nb + ty + j * 8;
        out[(long)n * K + kb + tx] = (__bf16)tile[tx][ty + j * 8];
    }
}

// ---------------------------------------------------------------------------
// Embedding: X[t*B + b][h] = emb[inputs[b][t]][h]  (bf16, time-major rows)
// ---------------------------------------------------------------------------
__global__ void embed_kernel(const int* __restrict__ inputs,
                             const float* __restrict__ emb,
                             __bf16* __restrict__ X) {
    int row = blockIdx.x;          // row = t*BATCH + b
    int t = row >> 5;
    int b = row & 31;
    int tok = inputs[b * T_IN + t];
    int h = threadIdx.x * 2;
    const float2 v = *(const float2*)(emb + (long)tok * HID + h);
    X[(long)row * HID + h]     = (__bf16)v.x;
    X[(long)row * HID + h + 1] = (__bf16)v.y;
}

// ---------------------------------------------------------------------------
// Tiled bf16 WMMA GEMM:  C[M,N] = A[M,K] * Bt[N,K]^T + bias[N]
// Block: 256 threads (8 waves), tile 64(M) x 128(N), K step 32.
// Double-buffered LDS staging; each wave computes a 32x32 patch (4 WMMA).
// No store-side row guard: caller guarantees C has >= ceil(M/64)*64 rows of
// backing storage (only the M=32 case over-writes, into workspace slack).
// ---------------------------------------------------------------------------
#define BM 64
#define BN 128
#define BK 32
#define LDP (BK + 16)   // padded LDS row stride (elems); 96B, 16B-aligned

template <bool OUT_BF16, bool RELU>
__global__ __launch_bounds__(256) void gemm_bf16_wmma(
    const __bf16* __restrict__ A, const __bf16* __restrict__ Bt,
    const float* __restrict__ bias, void* __restrict__ Cout,
    int M, int N, int K) {

    __shared__ __bf16 sA[2][BM * LDP];
    __shared__ __bf16 sB[2][BN * LDP];

    const int tid   = threadIdx.x;
    const int lane  = tid & 31;
    const int w     = tid >> 5;          // 0..7
    const int wm    = (w >> 2) * 32;     // 0 or 32
    const int wn    = (w & 3) * 32;      // 0,32,64,96
    const int bm    = blockIdx.y * BM;
    const int bn    = blockIdx.x * BN;
    const int lhalf = (lane >> 4);       // 0 or 1

    // staging addresses (fixed per thread)
    const int ra = tid >> 2;             // 0..63
    const int ca = (tid & 3) * 8;        // 0,8,16,24
    int rowA = bm + ra; if (rowA >= M) rowA = M - 1;   // read clamp only
    const __bf16* gA = A + (long)rowA * K + ca;
    const int rb = tid >> 1;             // 0..127
    const int cb = (tid & 1) * 16;       // 0,16
    const __bf16* gB = Bt + (long)(bn + rb) * K + cb;

    v8f acc[2][2] = {};

    // prologue: stage tile 0 into buffer 0
    {
        v8bf a0 = *(const v8bf*)(gA);
        v8bf b0 = *(const v8bf*)(gB);
        v8bf b1 = *(const v8bf*)(gB + 8);
        *(v8bf*)&sA[0][ra * LDP + ca]     = a0;
        *(v8bf*)&sB[0][rb * LDP + cb]     = b0;
        *(v8bf*)&sB[0][rb * LDP + cb + 8] = b1;
    }
    __syncthreads();

    const int nk = K / BK;
    for (int i = 0; i < nk; ++i) {
        const int buf = i & 1;
        const bool more = (i + 1 < nk);
        v8bf na = {}, nb0 = {}, nb1 = {};
        if (more) {                       // issue next tile's global loads early
            int kk = (i + 1) * BK;
            na  = *(const v8bf*)(gA + kk);
            nb0 = *(const v8bf*)(gB + kk);
            nb1 = *(const v8bf*)(gB + kk + 8);
        }

        // fragments from current buffer
        v16bf afrag[2], bfrag[2];
        #pragma unroll
        for (int mt = 0; mt < 2; ++mt) {
            int row = wm + mt * 16 + (lane & 15);
            int kb2 = lhalf * 16;
            v8bf lo = *(const v8bf*)&sA[buf][row * LDP + kb2];
            v8bf hi = *(const v8bf*)&sA[buf][row * LDP + kb2 + 8];
            v16bf a;
            #pragma unroll
            for (int q = 0; q < 8; ++q) { a[q] = lo[q]; a[q + 8] = hi[q]; }
            afrag[mt] = a;
        }
        #pragma unroll
        for (int nt = 0; nt < 2; ++nt) {
            int row = wn + nt * 16 + (lane & 15);
            int kb2 = lhalf * 16;
            v8bf lo = *(const v8bf*)&sB[buf][row * LDP + kb2];
            v8bf hi = *(const v8bf*)&sB[buf][row * LDP + kb2 + 8];
            v16bf b;
            #pragma unroll
            for (int q = 0; q < 8; ++q) { b[q] = lo[q]; b[q + 8] = hi[q]; }
            bfrag[nt] = b;
        }
        #pragma unroll
        for (int mt = 0; mt < 2; ++mt)
            #pragma unroll
            for (int nt = 0; nt < 2; ++nt)
                acc[mt][nt] = wmma_bf16(afrag[mt], bfrag[nt], acc[mt][nt]);

        if (more) {                       // stage next tile into other buffer
            *(v8bf*)&sA[buf ^ 1][ra * LDP + ca]     = na;
            *(v8bf*)&sB[buf ^ 1][rb * LDP + cb]     = nb0;
            *(v8bf*)&sB[buf ^ 1][rb * LDP + cb + 8] = nb1;
        }
        __syncthreads();
    }

    // epilogue: 32-bit addressing, compile-time specialized, no row guard
    float*  Cf = (float*)Cout;
    __bf16* Cb = (__bf16*)Cout;
    #pragma unroll
    for (int mt = 0; mt < 2; ++mt) {
        #pragma unroll
        for (int nt = 0; nt < 2; ++nt) {
            unsigned col  = bn + wn + nt * 16 + (lane & 15);
            float bv      = bias[col];
            unsigned row0 = bm + wm + mt * 16 + lhalf * 8;
            unsigned off  = row0 * (unsigned)N + col;
            #pragma unroll
            for (int e = 0; e < 8; ++e) {
                float v = acc[mt][nt][e] + bv;
                if (RELU) v = fmaxf(v, 0.0f);
                if (OUT_BF16) Cb[off] = (__bf16)v;
                else          Cf[off] = v;
                off += (unsigned)N;
            }
        }
    }
}

// ---------------------------------------------------------------------------
// Persistent GRU scan: one workgroup (32 waves), h resident in LDS (bf16).
// Per step: rec = h @ U + br via WMMA (M=32, N=1536, K=512), then gates.
// Wave w: m-tile (w&1), column group g=(w>>1): owns cols j in [32g,32g+32)
// for all three gates (z,r,h) so gate math stays in-register.
// ---------------------------------------------------------------------------
__global__ __launch_bounds__(1024) void gru_scan_kernel(
    const __bf16* __restrict__ Ut,     // [1536, 512] bf16 (U transposed)
    const float*  __restrict__ br,     // [1536]
    const float*  __restrict__ xp,     // x-projection (+bi) base
    long xp_stride,                    // elems per step (0 => constant input)
    __bf16* __restrict__ hs,           // [B, T, H] bf16 or null
    __bf16* __restrict__ hfinal,       // [B, H] bf16 or null
    int T) {

    __shared__ __bf16 h_lds[BATCH * HID];   // 32 KB

    const int tid   = threadIdx.x;
    const int lane  = tid & 31;
    const int w     = tid >> 5;             // 0..31
    const int m0    = (w & 1) * 16;
    const int jb    = (w >> 1) * 32;
    const int lhalf = (lane >> 4);

    for (int i = tid; i < BATCH * HID; i += 1024) h_lds[i] = (__bf16)0.0f;
    __syncthreads();

    for (int t = 0; t < T; ++t) {
        v8f acc[3][2] = {};   // [gate z/r/h][col subtile]

        // prefetch next step's x-projection rows (encoder path)
        if (xp_stride != 0 && t + 1 < T) {
            const float* nxt = xp + (long)(t + 1) * xp_stride + tid * 192;
            __builtin_prefetch(nxt, 0, 0);
        }

        for (int kk = 0; kk < HID; kk += 32) {
            // A fragment from LDS-resident h
            int rowm = m0 + (lane & 15);
            int kba  = kk + lhalf * 16;
            v8bf alo = *(const v8bf*)&h_lds[rowm * HID + kba];
            v8bf ahi = *(const v8bf*)&h_lds[rowm * HID + kba + 8];
            v16bf a;
            #pragma unroll
            for (int q = 0; q < 8; ++q) { a[q] = alo[q]; a[q + 8] = ahi[q]; }

            #pragma unroll
            for (int gate = 0; gate < 3; ++gate) {
                #pragma unroll
                for (int cc = 0; cc < 2; ++cc) {
                    int n0 = gate * HID + jb + cc * 16;
                    const __bf16* src = Ut + (long)(n0 + (lane & 15)) * HID
                                           + kk + lhalf * 16;
                    v8bf blo = *(const v8bf*)(src);
                    v8bf bhi = *(const v8bf*)(src + 8);
                    v16bf b;
                    #pragma unroll
                    for (int q = 0; q < 8; ++q) { b[q] = blo[q]; b[q + 8] = bhi[q]; }
                    acc[gate][cc] = wmma_bf16(a, b, acc[gate][cc]);
                }
            }
        }

        // --- gate math (all operands owned by this wave) ---
        const float* xpt = xp + (long)t * xp_stride;
        float hn[2][8];
        #pragma unroll
        for (int cc = 0; cc < 2; ++cc) {
            int col = jb + cc * 16 + (lane & 15);
            float brz = br[col];
            float brr = br[col + HID];
            float brh = br[col + 2 * HID];
            #pragma unroll
            for (int e = 0; e < 8; ++e) {
                int b = m0 + e + lhalf * 8;
                const float* x = xpt + (long)b * (3 * HID) + col;
                float z  = sigmoidf_fast(x[0]       + acc[0][cc][e] + brz);
                float r  = sigmoidf_fast(x[HID]     + acc[1][cc][e] + brr);
                float hh = tanhf(x[2 * HID] + r * (acc[2][cc][e] + brh));
                float hold = (float)h_lds[b * HID + col];
                hn[cc][e] = z * hold + (1.0f - z) * hh;
            }
        }
        __syncthreads();   // all reads of old h complete

        #pragma unroll
        for (int cc = 0; cc < 2; ++cc) {
            int col = jb + cc * 16 + (lane & 15);
            #pragma unroll
            for (int e = 0; e < 8; ++e) {
                int b = m0 + e + lhalf * 8;
                __bf16 hv = (__bf16)hn[cc][e];
                h_lds[b * HID + col] = hv;
                if (hs) hs[((long)b * T + t) * HID + col] = hv;
            }
        }
        __syncthreads();   // new h visible for next step
    }

    if (hfinal) {
        #pragma unroll
        for (int cc = 0; cc < 2; ++cc) {
            int col = jb + cc * 16 + (lane & 15);
            #pragma unroll
            for (int e = 0; e < 8; ++e) {
                int b = m0 + e + lhalf * 8;
                hfinal[b * HID + col] = h_lds[b * HID + col];
            }
        }
    }
}

// ---------------------------------------------------------------------------
// In-place row softmax over N columns, one block per row.
// ---------------------------------------------------------------------------
__global__ __launch_bounds__(256) void softmax_rows(float* __restrict__ out, int N) {
    __shared__ float red[256];
    long base = (long)blockIdx.x * N;
    int tid = threadIdx.x;

    float m = -INFINITY;
    for (int i = tid; i < N; i += 256) m = fmaxf(m, out[base + i]);
    red[tid] = m; __syncthreads();
    for (int s = 128; s > 0; s >>= 1) {
        if (tid < s) red[tid] = fmaxf(red[tid], red[tid + s]);
        __syncthreads();
    }
    m = red[0]; __syncthreads();

    float sum = 0.0f;
    for (int i = tid; i < N; i += 256) sum += __expf(out[base + i] - m);
    red[tid] = sum; __syncthreads();
    for (int s = 128; s > 0; s >>= 1) {
        if (tid < s) red[tid] += red[tid + s];
        __syncthreads();
    }
    float inv = 1.0f / red[0];
    for (int i = tid; i < N; i += 256)
        out[base + i] = __expf(out[base + i] - m) * inv;
}

// ---------------------------------------------------------------------------
// Launch
// ---------------------------------------------------------------------------
extern "C" void kernel_launch(void* const* d_in, const int* in_sizes, int n_in,
                              void* d_out, int out_size, void* d_ws, size_t ws_size,
                              hipStream_t stream) {
    (void)in_sizes; (void)n_in; (void)out_size; (void)ws_size;

    const int*   inputs = (const int*)  d_in[0];
    const float* emb    = (const float*)d_in[1];
    const float* enc_W  = (const float*)d_in[2];
    const float* enc_U  = (const float*)d_in[3];
    const float* enc_bi = (const float*)d_in[4];
    const float* enc_br = (const float*)d_in[5];
    const float* dec_W  = (const float*)d_in[6];
    const float* dec_U  = (const float*)d_in[7];
    const float* dec_bi = (const float*)d_in[8];
    const float* dec_br = (const float*)d_in[9];
    const float* w_relu = (const float*)d_in[10];
    const float* b_relu = (const float*)d_in[11];
    const float* w_out  = (const float*)d_in[12];
    const float* b_out  = (const float*)d_in[13];
    float* out = (float*)d_out;

    // Workspace carve-up (256B aligned)
    char* ws = (char*)d_ws;
    size_t off = 0;
    auto carve = [&](size_t bytes) -> void* {
        off = (off + 255) & ~(size_t)255;
        void* p = ws + off;
        off += bytes;
        return p;
    };
    const long MR = (long)T_IN * BATCH;        // 4096 rows
    __bf16* Xbf    = (__bf16*)carve(MR * HID * 2);
    __bf16* WencT  = (__bf16*)carve((long)3 * HID * HID * 2);
    __bf16* UencT  = (__bf16*)carve((long)3 * HID * HID * 2);
    __bf16* WdecT  = (__bf16*)carve((long)3 * HID * HID * 2);
    __bf16* UdecT  = (__bf16*)carve((long)3 * HID * HID * 2);
    __bf16* wreluT = (__bf16*)carve((long)HID * HID * 2);
    __bf16* woutT  = (__bf16*)carve((long)VOCAB * HID * 2);
    float*  xpEnc  = (float*) carve(MR * 3 * HID * 4);
    float*  xpDec  = (float*) carve((long)BM * 3 * HID * 4);  // 64-row slack (M=32 GEMM, no store guard)
    __bf16* hEnc   = (__bf16*)carve((long)BATCH * HID * 2);
    __bf16* hsBuf  = (__bf16*)carve(MR * HID * 2);
    __bf16* h1Buf  = (__bf16*)carve(MR * HID * 2);

    // 1) weight prep: fp32 [K,N] -> bf16 [N,K] (tiled, coalesced both sides)
    transpose_to_bf16<<<dim3(3 * HID / 32, HID / 32), dim3(32, 8), 0, stream>>>(enc_W, WencT, HID, 3 * HID);
    transpose_to_bf16<<<dim3(3 * HID / 32, HID / 32), dim3(32, 8), 0, stream>>>(enc_U, UencT, HID, 3 * HID);
    transpose_to_bf16<<<dim3(3 * HID / 32, HID / 32), dim3(32, 8), 0, stream>>>(dec_W, WdecT, HID, 3 * HID);
    transpose_to_bf16<<<dim3(3 * HID / 32, HID / 32), dim3(32, 8), 0, stream>>>(dec_U, UdecT, HID, 3 * HID);
    transpose_to_bf16<<<dim3(HID / 32, HID / 32),     dim3(32, 8), 0, stream>>>(w_relu, wreluT, HID, HID);
    transpose_to_bf16<<<dim3(VOCAB / 32, HID / 32),   dim3(32, 8), 0, stream>>>(w_out, woutT, HID, VOCAB);

    // 2) embedding lookup -> time-major bf16 X
    embed_kernel<<<(int)MR, 256, 0, stream>>>(inputs, emb, Xbf);

    // 3) encoder x-projection: xpEnc = X @ enc_W + enc_bi   [4096,1536]
    gemm_bf16_wmma<false, false><<<dim3(3 * HID / BN, (int)MR / BM), 256, 0, stream>>>(
        Xbf, WencT, enc_bi, xpEnc, (int)MR, 3 * HID, HID);

    // 4) encoder scan -> hEnc
    gru_scan_kernel<<<1, 1024, 0, stream>>>(
        UencT, enc_br, xpEnc, (long)BATCH * 3 * HID, nullptr, hEnc, T_IN);

    // 5) decoder x-projection (constant input): xpDec = hEnc @ dec_W + dec_bi [32,1536]
    gemm_bf16_wmma<false, false><<<dim3(3 * HID / BN, 1), 256, 0, stream>>>(
        hEnc, WdecT, dec_bi, xpDec, BATCH, 3 * HID, HID);

    // 6) decoder scan -> hs [B, T_OUT, H] (row = b*T_OUT + t)
    gru_scan_kernel<<<1, 1024, 0, stream>>>(
        UdecT, dec_br, xpDec, 0L, hsBuf, nullptr, T_OUT);

    // 7) h1 = relu(hs @ w_relu + b_relu)  (bf16 out)
    gemm_bf16_wmma<true, true><<<dim3(HID / BN, (int)MR / BM), 256, 0, stream>>>(
        hsBuf, wreluT, b_relu, h1Buf, (int)MR, HID, HID);

    // 8) logits = h1 @ w_out + b_out -> d_out (fp32)
    gemm_bf16_wmma<false, false><<<dim3(VOCAB / BN, (int)MR / BM), 256, 0, stream>>>(
        h1Buf, woutT, b_out, out, (int)MR, VOCAB, HID);

    // 9) in-place softmax over vocab
    softmax_rows<<<(int)MR, 256, 0, stream>>>(out, VOCAB);
}
